// NTXentLoss_35613868818747
// MI455X (gfx1250) — matrix operand order repaired and checked by reading, not compiled, not run on hardware
//
#include <hip/hip_runtime.h>

typedef __attribute__((ext_vector_type(16))) _Float16 v16h;
typedef __attribute__((ext_vector_type(8)))  _Float16 v8h;
typedef __attribute__((ext_vector_type(8)))  float    v8f;
typedef __attribute__((ext_vector_type(4)))  float    v4f;
typedef __attribute__((ext_vector_type(4)))  int      v4i;

#define BATCH   4096
#define NROWS   8192              // 2*BATCH
#define DIM     256
#define NT      (NROWS / 16)      // 512 column tiles
#define CCH     8                 // column chunks (parallelism over N)
#define TPC     (NT / CCH)        // 64 tiles per chunk
#define INV_T   2.0f              // 1/0.5
#define EPS_N   1e-8f
#define LDS_STRIDE 264            // halves per LDS row (256 + 8 pad; bank step 4)

union V16U { v16h v; v8h h[2]; };

#if __has_builtin(__builtin_amdgcn_global_load_async_to_lds_b128)
#define USE_ASYNC_LDS 1
typedef __attribute__((address_space(1))) v4i gas_v4i;   // global int4
typedef __attribute__((address_space(3))) v4i las_v4i;   // LDS int4
#define AS1_V4I(p) ((gas_v4i*)(p))
#define AS3_V4I(p) ((las_v4i*)(p))
#else
#define USE_ASYNC_LDS 0
#endif

__device__ __forceinline__ void wait_async_done() {
#if __has_builtin(__builtin_amdgcn_s_wait_asynccnt)
    __builtin_amdgcn_s_wait_asynccnt(0);
#else
    asm volatile("s_wait_asynccnt 0x0" ::: "memory");
#endif
}

// ---------------------------------------------------------------------------
// Kernel 1: row-normalize z = [z_i; z_j] -> f16 zn in workspace; zero d_out.
// ---------------------------------------------------------------------------
__global__ __launch_bounds__(256)
void ntxent_normalize_kernel(const float* __restrict__ z_i,
                             const float* __restrict__ z_j,
                             _Float16* __restrict__ zn,
                             float* __restrict__ out)
{
    if (blockIdx.x == 0 && threadIdx.x == 0) out[0] = 0.0f;

    const int lane = threadIdx.x & 31;
    const int row  = blockIdx.x * (blockDim.x >> 5) + (threadIdx.x >> 5);
    if (row >= NROWS) return;

    const float* src = (row < BATCH) ? (z_i + (size_t)row * DIM)
                                     : (z_j + (size_t)(row - BATCH) * DIM);
    v4f x0 = *(const v4f*)(src + lane * 8);
    v4f x1 = *(const v4f*)(src + lane * 8 + 4);

    float ss = x0.x*x0.x + x0.y*x0.y + x0.z*x0.z + x0.w*x0.w
             + x1.x*x1.x + x1.y*x1.y + x1.z*x1.z + x1.w*x1.w;
    #pragma unroll
    for (int m = 1; m <= 16; m <<= 1)
        ss += __shfl_xor(ss, m, 32);

    const float inv = 1.0f / fmaxf(sqrtf(ss), EPS_N);

    v8h o;
    o[0] = (_Float16)(x0.x * inv); o[1] = (_Float16)(x0.y * inv);
    o[2] = (_Float16)(x0.z * inv); o[3] = (_Float16)(x0.w * inv);
    o[4] = (_Float16)(x1.x * inv); o[5] = (_Float16)(x1.y * inv);
    o[6] = (_Float16)(x1.z * inv); o[7] = (_Float16)(x1.w * inv);
    *(v8h*)(zn + (size_t)row * DIM + lane * 8) = o;
}

// ---------------------------------------------------------------------------
// Kernel 2: per (m-tile, column-chunk): partial Sum_j exp(sim) (diag masked)
// + positive capture.  grid = (64 m-groups, 8 col-chunks), 8 waves/block.
// B tiles staged in LDS via double-buffered async copy; WMMA f16 compute.
// Writes E[chunk][row] (exactly once) and P[row] (owning chunk only).
// ---------------------------------------------------------------------------
__global__ __launch_bounds__(256)
void ntxent_wmma_partial_kernel(const _Float16* __restrict__ zn,
                                float* __restrict__ E,
                                float* __restrict__ P)
{
    __shared__ _Float16 lb[2][16 * LDS_STRIDE];   // 2 x 8.25 KB

    const int tid      = threadIdx.x;
    const int lane     = tid & 31;
    const int l16      = lane & 15;
    const int half_sel = lane >> 4;
    const int wid      = tid >> 5;                         // wave 0..7 in block
    const int m_base   = (blockIdx.x * 8 + wid) * 16;      // this wave's rows
    const int nt0      = blockIdx.y * TPC;                 // first column tile

    // ---- A slab: 16 rows x K=256 in 8 fragments (ISA 16-bit A layout).
    v16h a[8];
    {
        const _Float16* rp = zn + (size_t)(m_base + l16) * DIM;
        #pragma unroll
        for (int kf = 0; kf < 8; ++kf) {
            const int k0 = kf * 32;
            V16U t;
            t.h[0] = *(const v8h*)(rp + k0 + (half_sel ? 8  : 0));
            t.h[1] = *(const v8h*)(rp + k0 + (half_sel ? 24 : 16));
            a[kf] = t.v;
        }
    }

    // ---- Per-thread copy slice: 32 contiguous bytes of the 8 KB B tile.
    const int crow = (tid * 2) >> 5;               // tile row 0..15
    const int coff = ((tid * 2) & 31) * 8;         // half offset in row
    const size_t gso = (size_t)crow * DIM + coff;  // global halves from tile base
    const int    ldo = crow * LDS_STRIDE + coff;   // LDS halves from buffer base

#if !USE_ASYNC_LDS
    v8h p0, p1;                                    // register staging (fallback)
#endif

    // prefetch first tile into buffer 0
    {
        const _Float16* g = zn + (size_t)nt0 * 16 * DIM + gso;
#if USE_ASYNC_LDS
        _Float16* l = &lb[0][ldo];
        __builtin_amdgcn_global_load_async_to_lds_b128(AS1_V4I(g), AS3_V4I(l), 0, 0);
        __builtin_amdgcn_global_load_async_to_lds_b128(AS1_V4I(g), AS3_V4I(l), 16, 0);
#else
        p0 = *(const v8h*)(g);
        p1 = *(const v8h*)(g + 8);
#endif
    }

    v8f esum = {0.f,0.f,0.f,0.f,0.f,0.f,0.f,0.f};
    v8f posv = {0.f,0.f,0.f,0.f,0.f,0.f,0.f,0.f};

    for (int it = 0; it < TPC; ++it) {
        const int nt  = nt0 + it;
        const int cur = it & 1;

        // complete this thread's copy of buf[cur]
#if USE_ASYNC_LDS
        wait_async_done();
#else
        *(v8h*)(&lb[cur][ldo])     = p0;
        *(v8h*)(&lb[cur][ldo + 8]) = p1;
#endif
        __syncthreads();   // buf[cur] populated; everyone done reading buf[cur^1]

        // prefetch tile nt+1 into the other buffer (overlaps with compute)
        if (it + 1 < TPC) {
            const _Float16* g = zn + (size_t)(nt + 1) * 16 * DIM + gso;
#if USE_ASYNC_LDS
            _Float16* l = &lb[cur ^ 1][ldo];
            __builtin_amdgcn_global_load_async_to_lds_b128(AS1_V4I(g), AS3_V4I(l), 0, 0);
            __builtin_amdgcn_global_load_async_to_lds_b128(AS1_V4I(g), AS3_V4I(l), 16, 0);
#else
            p0 = *(const v8h*)(g);
            p1 = *(const v8h*)(g + 8);
#endif
        }

        // ---- hoist ALL B fragments (one ds_load clause), then WMMA chains
        const _Float16* bp = &lb[cur][l16 * LDS_STRIDE + half_sel * 8];
        v16h b[8];
        #pragma unroll
        for (int kf = 0; kf < 8; ++kf) {
            V16U t;
            t.h[0] = *(const v8h*)(bp + kf * 32);
            t.h[1] = *(const v8h*)(bp + kf * 32 + 16);
            b[kf] = t.v;
        }
        v8f cA = {0.f,0.f,0.f,0.f,0.f,0.f,0.f,0.f};
        v8f cB = {0.f,0.f,0.f,0.f,0.f,0.f,0.f,0.f};
        #pragma unroll
        for (int kf = 0; kf < 4; ++kf)
            cA = __builtin_amdgcn_wmma_f32_16x16x32_f16(
                    false, a[kf], false, b[kf], (short)0, cA, false, false);
        #pragma unroll
        for (int kf = 4; kf < 8; ++kf)
            cB = __builtin_amdgcn_wmma_f32_16x16x32_f16(
                    false, a[kf], false, b[kf], (short)0, cB, false, false);

        // ---- epilogue: scale, mask diagonal, capture positive, exp-accumulate
        const int col = nt * 16 + l16;
        #pragma unroll
        for (int v = 0; v < 8; ++v) {
            const int row = m_base + half_sel * 8 + v;
            const int pos = (row < BATCH) ? (row + BATCH) : (row - BATCH);
            const float s = (cA[v] + cB[v]) * INV_T;
            esum[v] += (col == row) ? 0.0f : __expf(s);
            if (col == pos) posv[v] = s;
        }
    }

    // reduce across the 16 lanes sharing the same rows (keep M-halves separate)
    #pragma unroll
    for (int m = 1; m <= 8; m <<= 1) {
        #pragma unroll
        for (int v = 0; v < 8; ++v) {
            esum[v] += __shfl_xor(esum[v], m, 32);
            posv[v] += __shfl_xor(posv[v], m, 32);   // exactly one lane nonzero
        }
    }

    // write partials: E[chunk][row] once per (row, chunk); P[row] by owner chunk
    if (l16 == 0) {   // lanes 0 (rows m..m+7) and 16 (rows m+8..m+15)
        const int rbase = m_base + half_sel * 8;
        float* Ec = E + (size_t)blockIdx.y * NROWS;
        #pragma unroll
        for (int v = 0; v < 8; ++v)
            Ec[rbase + v] = esum[v];

        // pos tile for the whole 16-row stripe (BATCH is tile-aligned)
        const int pos_tile = (m_base >> 4) + ((m_base < BATCH) ? (BATCH / 16)
                                                              : -(BATCH / 16));
        if ((pos_tile / TPC) == (int)blockIdx.y) {
            #pragma unroll
            for (int v = 0; v < 8; ++v)
                P[rbase + v] = posv[v];
        }
    }
}

// ---------------------------------------------------------------------------
// Kernel 3: loss = mean_r [ log(sum_c E[c][r]) - P[r] ]
// ---------------------------------------------------------------------------
__global__ __launch_bounds__(256)
void ntxent_finalize_kernel(const float* __restrict__ E,
                            const float* __restrict__ P,
                            float* __restrict__ out)
{
    const int r = blockIdx.x * blockDim.x + threadIdx.x;   // 0..8191
    float e = 0.0f;
    #pragma unroll
    for (int c = 0; c < CCH; ++c)
        e += E[(size_t)c * NROWS + r];
    float v = __logf(e) - P[r];
    #pragma unroll
    for (int m = 1; m <= 16; m <<= 1)
        v += __shfl_xor(v, m, 32);
    if ((threadIdx.x & 31) == 0)
        atomicAdd(out, v * (1.0f / (float)NROWS));
}

// ---------------------------------------------------------------------------
extern "C" void kernel_launch(void* const* d_in, const int* in_sizes, int n_in,
                              void* d_out, int out_size, void* d_ws, size_t ws_size,
                              hipStream_t stream)
{
    const float* z_i = (const float*)d_in[0];
    const float* z_j = (const float*)d_in[1];
    float* out = (float*)d_out;

    // workspace layout: zn (4 MB f16) | E (CCH*NROWS f32, 256 KB) | P (32 KB)
    _Float16* zn = (_Float16*)d_ws;
    float* E = (float*)((char*)d_ws + (size_t)NROWS * DIM * sizeof(_Float16));
    float* P = E + (size_t)CCH * NROWS;

    ntxent_normalize_kernel<<<NROWS / 8, 256, 0, stream>>>(z_i, z_j, zn, out);
    ntxent_wmma_partial_kernel<<<dim3(NROWS / (8 * 16), CCH), 256, 0, stream>>>(zn, E, P);
    ntxent_finalize_kernel<<<NROWS / 256, 256, 0, stream>>>(E, P, out);
}
